// SurfGVP_77446850281848
// MI455X (gfx1250) — compile-verified, matrix operand order checked by reading, not run on hardware
//
#include <hip/hip_runtime.h>
#include <hip/hip_bf16.h>
#include <math.h>

typedef _Float16 f16;
typedef __attribute__((ext_vector_type(16))) _Float16 v16h;
typedef __attribute__((ext_vector_type(8)))  _Float16 f16x8;
typedef __attribute__((ext_vector_type(8)))  float    v8f;

#define CDIV(a,b) (((a)+(b)-1)/(b))

// ---------------- constants (mirror reference) ----------------
#define DIN   128
#define SDIN  16
#define NS    128
#define NVv   16
#define ES    32
#define EV    1
#define RBF_DIM 16
#define N_RES  2048
#define N_SURF 8192
#define E_RES  61440
#define E_SURF 131072

// =====================================================================
// WMMA GEMM: C = act(A(f16, MxKp, lda=Kp) @ Bt(f16, NxKp transposed) + bias)
// Requirements (guaranteed by host): M % 64 == 0, N % 16 == 0, Kp % 32 == 0,
// A and Bt zero-padded to Kp. One wave per 16x16 C tile, 4 waves/block.
// Inner loop: 4x global_load_b128 + 1x v_wmma_f32_16x16x32_f16, unpredicated.
// =====================================================================
__global__ __launch_bounds__(128)
void k_gemm(const f16* __restrict__ A, int lda,
            const f16* __restrict__ Bt, int ldbt,
            const float* __restrict__ bias,
            float* __restrict__ C, f16* __restrict__ C16, int ldc,
            int M, int N, int Kp, int act)
{
  int lane = threadIdx.x & 31;
  int wave = threadIdx.x >> 5;
  int halfw = lane >> 4;
  int lr   = lane & 15;
  int m0 = (blockIdx.y * 4 + wave) * 16;
  int n0 = blockIdx.x * 16;
  int q  = halfw * 8;           // K-subgroup offset per ISA 16-bit operand layout
  const f16* pa = A  + (size_t)(m0 + lr) * lda  + q;
  const f16* pb = Bt + (size_t)(n0 + lr) * ldbt + q;
  v8f acc = {};
  union AB { v16h v; struct { f16x8 lo, hi; } p; };
  for (int k0 = 0; k0 < Kp; k0 += 32) {
    AB ua, ub;
    ua.p.lo = *reinterpret_cast<const f16x8*>(pa + k0);
    ua.p.hi = *reinterpret_cast<const f16x8*>(pa + k0 + 16);
    ub.p.lo = *reinterpret_cast<const f16x8*>(pb + k0);
    ub.p.hi = *reinterpret_cast<const f16x8*>(pb + k0 + 16);
    acc = __builtin_amdgcn_wmma_f32_16x16x32_f16(false, ua.v, false, ub.v,
                                                 (short)0, acc, false, false);
  }
  union { v8f v; float f[8]; } uc; uc.v = acc;
  int n = n0 + lr;
  float bval = bias ? bias[n] : 0.f;
#pragma unroll
  for (int i = 0; i < 8; ++i) {
    int m = m0 + i + halfw * 8;          // ISA C layout: vgpr i -> row i + 8*half
    float val = uc.f[i] + bval;
    if (act == 1) val = fmaxf(val, 0.f);
    C[(size_t)m * ldc + n] = val;
    if (C16) C16[(size_t)m * ldc + n] = (f16)val;
  }
}

// pack B (f32, KxN row-major) -> Bt (f16, NxKp, zero-padded)
__global__ void k_pack_Bt(const float* __restrict__ B, int K, int N,
                          f16* __restrict__ Bt, int Kp)
{
  long t = blockIdx.x * (long)blockDim.x + threadIdx.x;
  if (t >= (long)N * Kp) return;
  int n = (int)(t / Kp), k = (int)(t % Kp);
  Bt[t] = (k < K) ? (f16)B[(size_t)k * N + n] : (f16)0.f;
}

// ---------------- elementwise / helper kernels ----------------
__global__ void k_zero(float* p, long n) {
  long t = blockIdx.x * (long)blockDim.x + threadIdx.x;
  if (t < n) p[t] = 0.f;
}
__global__ void k_f32_to_f16(const float* x, f16* y, long n) {
  long t = blockIdx.x * (long)blockDim.x + threadIdx.x;
  if (t < n) y[t] = (f16)x[t];
}
__global__ void k_iota_div(int* o, int n, int d) {
  int t = blockIdx.x * blockDim.x + threadIdx.x;
  if (t < n) o[t] = t / d;
}
__global__ void k_count(const int* idx, int E, float* cnt) {
  int t = blockIdx.x * blockDim.x + threadIdx.x;
  if (t < E) atomicAdd(&cnt[idx[t]], 1.f);
}
__global__ void k_seg_add(const float* x, const int* idx, int E, int D, float* out) {
  long t = blockIdx.x * (long)blockDim.x + threadIdx.x;
  if (t >= (long)E * D) return;
  int r = (int)(t / D), d = (int)(t % D);
  atomicAdd(&out[(size_t)idx[r] * D + d], x[t]);
}

// residual + LayerNorm for s; vector-norm rescale for v (gvp_layernorm)
__global__ void k_add_ln(float* s, float* v, const float* ds, const float* dv,
                         const float* cnt, const float* g, const float* b,
                         int N, int D, int NV, int do_relu)
{
  int n = blockIdx.x * blockDim.x + threadIdx.x;
  if (n >= N) return;
  float inv = cnt ? (1.f / fmaxf(cnt[n], 1.f)) : 1.f;
  float buf[256];
  float* ps = s + (size_t)n * D;
  float mu = 0.f;
  for (int d = 0; d < D; ++d) {
    float val = ps[d] + (ds ? ds[(size_t)n * D + d] * inv : 0.f);
    buf[d] = val; mu += val;
  }
  mu /= D;
  float var = 0.f;
  for (int d = 0; d < D; ++d) { float e = buf[d] - mu; var += e * e; }
  var /= D;
  float rstd = rsqrtf(var + 1e-5f);
  for (int d = 0; d < D; ++d) {
    float val = (buf[d] - mu) * rstd * g[d] + b[d];
    if (do_relu) val = fmaxf(val, 0.f);
    ps[d] = val;
  }
  if (v) {
    float tv[48];
    float* pv = v + (size_t)n * NV * 3;
    float acc = 0.f;
    for (int h = 0; h < NV; ++h) {
      float x = pv[h*3+0] + (dv ? dv[((size_t)n*NV+h)*3+0] * inv : 0.f);
      float y = pv[h*3+1] + (dv ? dv[((size_t)n*NV+h)*3+1] * inv : 0.f);
      float z = pv[h*3+2] + (dv ? dv[((size_t)n*NV+h)*3+2] * inv : 0.f);
      tv[h*3]=x; tv[h*3+1]=y; tv[h*3+2]=z;
      acc += fmaxf(x*x+y*y+z*z, 1e-8f);
    }
    float is = rsqrtf(acc / NV);
    for (int h = 0; h < NV * 3; ++h) pv[h] = tv[h] * is;
  }
}

// edge geometry: dvec = pos[a]-pos[b]; rbf(16); ev
__global__ void k_edge_geom(const float* pos, const int* ai, const int* bi, int E,
                            float* es_raw, float* ev)
{
  int r = blockIdx.x * blockDim.x + threadIdx.x;
  if (r >= E) return;
  int a = ai[r], bb = bi[r];
  float dx = pos[3*a] - pos[3*bb], dy = pos[3*a+1] - pos[3*bb+1], dz = pos[3*a+2] - pos[3*bb+2];
  float d = sqrtf(fmaxf(dx*dx + dy*dy + dz*dz, 1e-12f));
  for (int j = 0; j < 16; ++j) {
    float mu = (20.f / 15.f) * j;
    float t = (d - mu) / 1.25f;
    es_raw[(size_t)r*16 + j] = expf(-t * t);
  }
  ev[3*r] = dx; ev[3*r+1] = dy; ev[3*r+2] = dz;
}

// fused gvp_layernorm(We_ln) + gvp(We) for edges: writes es_out(32), ev in-place
__global__ void k_gvp_edge_We(const float* es_raw, float* ev_io,
                              const float* ln_b, const float* ln_g,
                              const float* wh, const float* ws_b, const float* ws_w,
                              const float* wsv_b, const float* wsv_w, const float* wv,
                              int E, float* es_out)
{
  int r = blockIdx.x * blockDim.x + threadIdx.x;
  if (r >= E) return;
  const float* pe = es_raw + (size_t)r * 16;
  float lnv[16]; float mu = 0.f;
  for (int j = 0; j < 16; ++j) mu += pe[j];
  mu /= 16.f;
  float var = 0.f;
  for (int j = 0; j < 16; ++j) { float e = pe[j] - mu; var += e * e; }
  var /= 16.f;
  float rstd = rsqrtf(var + 1e-5f);
  for (int j = 0; j < 16; ++j) lnv[j] = (pe[j] - mu) * rstd * ln_g[j] + ln_b[j];
  float ex = ev_io[3*r], ey = ev_io[3*r+1], ez = ev_io[3*r+2];
  float n2 = fmaxf(ex*ex + ey*ey + ez*ez, 1e-8f);
  float is = rsqrtf(n2);
  ex *= is; ey *= is; ez *= is;
  float w0 = wh[0];
  float vx = ex * w0, vy = ey * w0, vz = ez * w0;
  float vn = sqrtf(fmaxf(vx*vx + vy*vy + vz*vz, 1e-8f));
  float gate = wsv_b[0];
  for (int o = 0; o < 32; ++o) {
    float acc = ws_b[o];
    for (int j = 0; j < 16; ++j) acc += lnv[j] * ws_w[j*32 + o];
    acc += vn * ws_w[16*32 + o];
    es_out[(size_t)r*32 + o] = acc;
    gate += acc * wsv_w[o];
  }
  float gg = 1.f / (1.f + expf(-gate));
  float wvv = wv[0] * gg;
  ev_io[3*r] = vx * wvv; ev_io[3*r+1] = vy * wvv; ev_io[3*r+2] = vz * wvv;
}

// vh for msg0: mv = [v[src](16), ev(1), v[dst](16)] -> vh(33,3) via wh(33,33)
__global__ void k_vh_msg0(const float* v, const float* ev, const int* src, const int* dst,
                          const float* wh, int E, float* vh)
{
  long t = blockIdx.x * (long)blockDim.x + threadIdx.x;
  if (t >= (long)E * 33) return;
  int r = (int)(t / 33), h = (int)(t % 33);
  int sI = src[r], dI = dst[r];
  float a0 = 0, a1 = 0, a2 = 0;
  for (int i = 0; i < 16; ++i) {
    float w = wh[i*33 + h]; const float* p = &v[((size_t)sI*16 + i) * 3];
    a0 += p[0]*w; a1 += p[1]*w; a2 += p[2]*w;
  }
  { float w = wh[16*33 + h]; const float* p = &ev[(size_t)r*3];
    a0 += p[0]*w; a1 += p[1]*w; a2 += p[2]*w; }
  for (int i = 0; i < 16; ++i) {
    float w = wh[(17+i)*33 + h]; const float* p = &v[((size_t)dI*16 + i) * 3];
    a0 += p[0]*w; a1 += p[1]*w; a2 += p[2]*w;
  }
  float* o = &vh[t * 3]; o[0]=a0; o[1]=a1; o[2]=a2;
}

// dense vh: vh[r][h][c] = sum_i vin[r][i][c] * wh[i][h]
__global__ void k_vh_dense(const float* vin, int R, int VI,
                           const float* wh, int H, float* vh)
{
  long t = blockIdx.x * (long)blockDim.x + threadIdx.x;
  if (t >= (long)R * H) return;
  int r = (int)(t / H), h = (int)(t % H);
  const float* p = vin + (size_t)r * VI * 3;
  float a0 = 0, a1 = 0, a2 = 0;
  for (int i = 0; i < VI; ++i) {
    float w = wh[i*H + h];
    a0 += p[i*3]*w; a1 += p[i*3+1]*w; a2 += p[i*3+2]*w;
  }
  float* o = &vh[t * 3]; o[0]=a0; o[1]=a1; o[2]=a2;
}

// msg0 A build: [s[src](128), es(32), s[dst](128), vn(33)] -> f16 row, lda=352 (pad 0)
__global__ void k_build_msg0A(const float* s, const float* es, const int* src, const int* dst,
                              const float* vh, int E, f16* A, int lda)
{
  int r = blockIdx.x * blockDim.x + threadIdx.x;
  if (r >= E) return;
  f16* row = A + (size_t)r * lda;
  const float* ps = s + (size_t)src[r] * 128;
  const float* pd = s + (size_t)dst[r] * 128;
  for (int i = 0; i < 128; ++i) row[i] = (f16)ps[i];
  const float* pe = es + (size_t)r * 32;
  for (int i = 0; i < 32; ++i) row[128+i] = (f16)pe[i];
  for (int i = 0; i < 128; ++i) row[160+i] = (f16)pd[i];
  const float* pv = vh + (size_t)r * 33 * 3;
  for (int h = 0; h < 33; ++h) {
    float x = pv[h*3], y = pv[h*3+1], z = pv[h*3+2];
    row[288+h] = (f16)sqrtf(fmaxf(x*x + y*y + z*z, 1e-8f));
  }
  for (int i = 321; i < lda; ++i) row[i] = (f16)0.f;
}

// A build: [X(D1), vn_from_vh(H)] -> f16 row (lda, zero-padded)
__global__ void k_build_cat_vn(const float* X, int D1, const float* vh, int H,
                               f16* A, int R, int lda)
{
  int r = blockIdx.x * blockDim.x + threadIdx.x;
  if (r >= R) return;
  f16* row = A + (size_t)r * lda;
  const float* px = X + (size_t)r * D1;
  for (int i = 0; i < D1; ++i) row[i] = (f16)px[i];
  const float* pv = vh + (size_t)r * H * 3;
  for (int h = 0; h < H; ++h) {
    float x = pv[h*3], y = pv[h*3+1], z = pv[h*3+2];
    row[D1+h] = (f16)sqrtf(fmaxf(x*x + y*y + z*z, 1e-8f));
  }
  for (int i = D1 + H; i < lda; ++i) row[i] = (f16)0.f;
}

// A build: [X1(D1), X2(D2)] -> f16 (lda, zero-padded)
__global__ void k_build_cat2(const float* X1, int D1, const float* X2, int D2,
                             f16* A, int R, int lda)
{
  int r = blockIdx.x * blockDim.x + threadIdx.x;
  if (r >= R) return;
  f16* row = A + (size_t)r * lda;
  const float* p1 = X1 + (size_t)r * D1;
  const float* p2 = X2 + (size_t)r * D2;
  for (int i = 0; i < D1; ++i) row[i] = (f16)p1[i];
  for (int i = 0; i < D2; ++i) row[D1+i] = (f16)p2[i];
  for (int i = D1 + D2; i < lda; ++i) row[i] = (f16)0.f;
}

// v_out = (vh @ wv) * sigmoid(gate)
__global__ void k_vout(const float* vh, int H, const float* wv, int O,
                       const float* gate, int R, float* out)
{
  long t = blockIdx.x * (long)blockDim.x + threadIdx.x;
  if (t >= (long)R * O) return;
  int r = (int)(t / O), o = (int)(t % O);
  const float* pv = vh + (size_t)r * H * 3;
  float a0 = 0, a1 = 0, a2 = 0;
  for (int h = 0; h < H; ++h) {
    float w = wv[h*O + o];
    a0 += pv[h*3]*w; a1 += pv[h*3+1]*w; a2 += pv[h*3+2]*w;
  }
  float gg = 1.f / (1.f + expf(-gate[t]));
  float* po = out + t * 3;
  po[0]=a0*gg; po[1]=a1*gg; po[2]=a2*gg;
}

// knn surf->res (top-3) + mean distance
__global__ void k_knn_res(const float* q, const float* rp, int Nq, int Nr,
                          int* idx, float* dbar)
{
  int i = blockIdx.x * blockDim.x + threadIdx.x;
  if (i >= Nq) return;
  float qx = q[3*i], qy = q[3*i+1], qz = q[3*i+2];
  float bd[3] = {3e38f, 3e38f, 3e38f}; int bix[3] = {0, 0, 0};
  for (int j = 0; j < Nr; ++j) {
    float dx = qx - rp[3*j], dy = qy - rp[3*j+1], dz = qz - rp[3*j+2];
    float d2 = dx*dx + dy*dy + dz*dz;
    if (d2 < bd[2]) {
      int p = 2;
      while (p > 0 && bd[p-1] > d2) { bd[p]=bd[p-1]; bix[p]=bix[p-1]; --p; }
      bd[p] = d2; bix[p] = j;
    }
  }
  idx[3*i]=bix[0]; idx[3*i+1]=bix[1]; idx[3*i+2]=bix[2];
  float s = 0.f;
  for (int k = 0; k < 3; ++k) s += sqrtf(fmaxf(bd[k], 1e-12f));
  dbar[i] = s / 3.f;
}

// knn surf->surf top-16 (exclude self), sorted ascending d2
__global__ void k_knn_surf(const float* q, int Nq, int* ssrc)
{
  int i = blockIdx.x * blockDim.x + threadIdx.x;
  if (i >= Nq) return;
  float qx = q[3*i], qy = q[3*i+1], qz = q[3*i+2];
  float bd[16]; int bix[16];
  for (int k = 0; k < 16; ++k) { bd[k] = 3e38f; bix[k] = 0; }
  for (int j = 0; j < Nq; ++j) {
    if (j == i) continue;
    float dx = qx - q[3*j], dy = qy - q[3*j+1], dz = qz - q[3*j+2];
    float d2 = dx*dx + dy*dy + dz*dz;
    if (d2 < bd[15]) {
      int p = 15;
      while (p > 0 && bd[p-1] > d2) { bd[p]=bd[p-1]; bix[p]=bix[p-1]; --p; }
      bd[p] = d2; bix[p] = j;
    }
  }
  for (int k = 0; k < 16; ++k) ssrc[(size_t)i*16 + k] = bix[k];
}

// A = mean_k input[surf2res[s][k]]  (f16)
__global__ void k_mean_gather(const float* x, const int* idx, f16* A, int Ns)
{
  long t = blockIdx.x * (long)blockDim.x + threadIdx.x;
  if (t >= (long)Ns * 128) return;
  int s = (int)(t / 128), d = (int)(t % 128);
  float acc = (x[(size_t)idx[3*s]*128 + d] + x[(size_t)idx[3*s+1]*128 + d] +
               x[(size_t)idx[3*s+2]*128 + d]) * (1.f / 3.f);
  A[t] = (f16)acc;
}
__global__ void k_add_dbar(float* h, const float* dbar, const float* wrow, int Ns)
{
  long t = blockIdx.x * (long)blockDim.x + threadIdx.x;
  if (t >= (long)Ns * 128) return;
  int s = (int)(t / 128), d = (int)(t % 128);
  h[t] += dbar[s] * wrow[d];
}

__global__ void k_final_nf(const float* bb, const float* so, const int* r2s,
                           float* out)
{
  long t = blockIdx.x * (long)blockDim.x + threadIdx.x;
  if (t >= (long)N_RES * 128) return;
  int i = (int)(t / 128), d = (int)(t % 128);
  float acc = bb[t];
  for (int k = 0; k < 8; ++k)
    acc += so[(size_t)r2s[8*i + k] * 128 + d] * 0.125f;
  out[128 + t] = acc;
}
__global__ void k_graph_sum(float* out)
{
  int d = threadIdx.x;
  if (d >= 128) return;
  float acc = 0.f;
  for (int i = 0; i < N_RES; ++i) acc += out[128 + (size_t)i*128 + d];
  out[d] = acc;
}

// ---------------- host-side param layout (jax pytree: dict keys sorted) ----------------
struct LnP { long b, g; };
struct GvpP { long wh, ws_b, ws_w, wsv_b, wsv_w, wv; int h; };
struct ConvP { GvpP ff0, ff1; LnP ln0, ln1; GvpP msg0, msg1, msg2; };

static long g_cur;
static long take(long n) { long o = g_cur; g_cur += n; return o; }
static LnP ln_take(int d) { LnP l; l.b = take(d); l.g = take(d); return l; }
static GvpP gvp_take(int si, int vi, int so, int vo) {
  GvpP g; g.wh = g.wsv_b = g.wsv_w = g.wv = -1; g.h = 0;
  if (vi) {
    g.h = vi > vo ? vi : vo;
    g.wh = take((long)vi * g.h);
    g.ws_b = take(so);
    g.ws_w = take((long)(si + g.h) * so);
    if (vo) { g.wsv_b = take(vo); g.wsv_w = take((long)so * vo); g.wv = take((long)g.h * vo); }
  } else {
    g.ws_b = take(so);
    g.ws_w = take((long)si * so);
  }
  return g;
}
static ConvP conv_take() {
  ConvP c;
  c.ff0 = gvp_take(NS, NVv, 4*NS, 2*NVv);
  c.ff1 = gvp_take(4*NS, 2*NVv, NS, NVv);
  c.ln0 = ln_take(NS); c.ln1 = ln_take(NS);
  c.msg0 = gvp_take(2*NS + ES, 2*NVv + EV, NS, NVv);
  c.msg1 = gvp_take(NS, NVv, NS, NVv);
  c.msg2 = gvp_take(NS, NVv, NS, NVv);
  return c;
}

extern "C" void kernel_launch(void* const* d_in, const int* in_sizes, int n_in,
                              void* d_out, int out_size, void* d_ws, size_t ws_size,
                              hipStream_t stream)
{
  const float* in_x  = (const float*)d_in[0];   // (2048,128)
  const float* npos  = (const float*)d_in[1];   // (2048,3)
  const float* spos  = (const float*)d_in[2];   // (8192,3)
  const float* sfeat = (const float*)d_in[3];   // (8192,16)
  const int*   eidx  = (const int*)d_in[4];     // (2,61440)
  const int*   r2s   = (const int*)d_in[5];     // (2048,8)
  const float* Pf    = (const float*)d_in[6];   // flattened params
  float* out = (float*)d_out;

  // ---- param offsets (sorted-key pytree order) ----
  g_cur = 0;
  GvpP We      = gvp_take(RBF_DIM, 1, ES, EV);
  LnP  We_ln   = ln_take(RBF_DIM);
  GvpP Wout    = gvp_take(NS, NVv, NS, 0);
  LnP  Wout_ln = ln_take(NS);
  GvpP Wv      = gvp_take(DIN, 0, NS, NVv);
  LnP  Wv_ln   = ln_take(DIN);
  ConvP layers[3]; for (int i = 0; i < 3; ++i) layers[i] = conv_take();
  long mlp_b1 = take(2*DIN);
  long mlp_b2 = take(DIN);
  LnP  mlp_ln = ln_take(2*DIN);
  long mlp_w1 = take((long)(DIN+SDIN)*2*DIN);
  long mlp_w2 = take((long)2*DIN*DIN);
  long res_emb = take((long)DIN*DIN);
  GvpP sWe      = gvp_take(RBF_DIM, 1, ES, EV);
  LnP  sWe_ln   = ln_take(RBF_DIM);
  GvpP sWout    = gvp_take(NS, NVv, NS, 0);
  LnP  sWout_ln = ln_take(NS);
  GvpP sWv      = gvp_take(DIN, 0, NS, NVv);
  LnP  sWv_ln   = ln_take(DIN);
  long surf_in_lin = take((long)(DIN+1)*DIN);
  ConvP slayers[3]; for (int i = 0; i < 3; ++i) slayers[i] = conv_take();

  auto pp = [&](long o) { return Pf + o; };

  // ---- workspace allocator ----
  char* wsp = (char*)d_ws;
  size_t wcur = 0;
  auto alloc = [&](size_t bytes) -> void* {
    size_t o = (wcur + 255) & ~(size_t)255; wcur = o + bytes; return (void*)(wsp + o);
  };
  float* s_res = (float*)alloc((size_t)N_RES*128*4);
  float* v_res = (float*)alloc((size_t)N_RES*48*4);
  float* hs_s  = (float*)alloc((size_t)N_SURF*128*4);
  float* hv_s  = (float*)alloc((size_t)N_SURF*48*4);
  float* es_r  = (float*)alloc((size_t)E_RES*32*4);
  float* ev_r  = (float*)alloc((size_t)E_RES*3*4);
  float* ses   = (float*)alloc((size_t)E_SURF*32*4);
  float* sev   = (float*)alloc((size_t)E_SURF*3*4);
  float* raw16 = (float*)alloc((size_t)E_SURF*16*4);
  int*   ssrc  = (int*)alloc((size_t)E_SURF*4);
  int*   sdst  = (int*)alloc((size_t)E_SURF*4);
  int*   s2r   = (int*)alloc((size_t)N_SURF*3*4);
  float* dbar  = (float*)alloc((size_t)N_SURF*4);
  float* cnt_r = (float*)alloc((size_t)N_RES*4);
  float* cnt_s = (float*)alloc((size_t)N_SURF*4);
  float* hbuf  = (float*)alloc((size_t)N_SURF*128*4);
  float* h1    = (float*)alloc((size_t)N_SURF*256*4);
  float* vh33  = (float*)alloc((size_t)E_SURF*33*3*4);
  float* vh16  = (float*)alloc((size_t)E_SURF*16*3*4);
  float* mv    = (float*)alloc((size_t)E_SURF*48*4);
  float* ms    = (float*)alloc((size_t)E_SURF*128*4);
  f16*   ms16  = (f16*)alloc((size_t)E_SURF*128*2);
  float* gateE = (float*)alloc((size_t)E_SURF*16*4);
  f16*   Acat  = (f16*)alloc((size_t)E_SURF*352*2);
  float* vhf   = (float*)alloc((size_t)N_SURF*32*3*4);
  float* fv    = (float*)alloc((size_t)N_SURF*32*3*4);
  float* fs    = (float*)alloc((size_t)N_SURF*512*4);
  f16*   fs16  = (f16*)alloc((size_t)N_SURF*512*2);
  float* gateN = (float*)alloc((size_t)N_SURF*32*4);
  f16*   fA    = (f16*)alloc((size_t)N_SURF*544*2);
  float* ag_s  = (float*)alloc((size_t)N_SURF*128*4);
  float* ag_v  = (float*)alloc((size_t)N_SURF*48*4);
  float* hs2   = (float*)alloc((size_t)N_SURF*128*4);
  f16*   hs2_16= (f16*)alloc((size_t)N_SURF*128*2);
  float* hv2   = (float*)alloc((size_t)N_SURF*48*4);
  float* bbuf  = (float*)alloc((size_t)N_RES*128*4);
  float* soutb = (float*)alloc((size_t)N_SURF*128*4);
  f16*   Btbuf = (f16*)alloc((size_t)96*1024*2);   // max N*Kp = 512*160

  auto launch1 = [&](long n, int bs) { return dim3((unsigned)CDIV(n, bs)); };

  // pack-B + WMMA GEMM (lda must equal roundup(K,32); M%64==0, N%16==0)
  auto gemm = [&](const f16* A, int lda, const float* B, const float* bias,
                  float* C, f16* C16, int ldc, int M, int Nn, int K, int act) {
    int Kp = (K + 31) & ~31;
    k_pack_Bt<<<launch1((long)Nn*Kp,256),256,0,stream>>>(B, K, Nn, Btbuf, Kp);
    dim3 grid(Nn/16, M/64);
    k_gemm<<<grid, 128, 0, stream>>>(A, lda, Btbuf, Kp, bias, C, C16, ldc, M, Nn, Kp, act);
  };

  const int* e_src = eidx;
  const int* e_dst = eidx + E_RES;

  // ================= residue node init =================
  k_f32_to_f16<<<launch1((long)N_RES*128,256),256,0,stream>>>(in_x, fA, (long)N_RES*128);
  gemm(fA, 128, pp(res_emb), nullptr, s_res /*h_s temp*/, nullptr, 128, N_RES, 128, 128, 0);
  k_add_ln<<<launch1(N_RES,128),128,0,stream>>>(s_res, nullptr, nullptr, nullptr, nullptr,
                                                pp(Wv_ln.g), pp(Wv_ln.b), N_RES, 128, 0, 0);
  k_f32_to_f16<<<launch1((long)N_RES*128,256),256,0,stream>>>(s_res, fA, (long)N_RES*128);
  gemm(fA, 128, pp(Wv.ws_w), pp(Wv.ws_b), s_res, nullptr, 128, N_RES, 128, 128, 0);
  k_zero<<<launch1((long)N_RES*48,256),256,0,stream>>>(v_res, (long)N_RES*48);

  // residue edges: dvec = pos[dst]-pos[src]
  k_edge_geom<<<launch1(E_RES,256),256,0,stream>>>(npos, e_dst, e_src, E_RES, raw16, ev_r);
  k_gvp_edge_We<<<launch1(E_RES,128),128,0,stream>>>(raw16, ev_r,
      pp(We_ln.b), pp(We_ln.g), pp(We.wh), pp(We.ws_b), pp(We.ws_w),
      pp(We.wsv_b), pp(We.wsv_w), pp(We.wv), E_RES, es_r);
  k_zero<<<launch1(N_RES,256),256,0,stream>>>(cnt_r, N_RES);
  k_count<<<launch1(E_RES,256),256,0,stream>>>(e_dst, E_RES, cnt_r);

  // ================= surface node init =================
  k_knn_res<<<launch1(N_SURF,128),128,0,stream>>>(spos, npos, N_SURF, N_RES, s2r, dbar);
  k_mean_gather<<<launch1((long)N_SURF*128,256),256,0,stream>>>(in_x, s2r, fA, N_SURF);
  gemm(fA, 128, pp(surf_in_lin), nullptr, hbuf, nullptr, 128, N_SURF, 128, 128, 0);
  k_add_dbar<<<launch1((long)N_SURF*128,256),256,0,stream>>>(hbuf, dbar, pp(surf_in_lin)+128*128, N_SURF);
  k_build_cat2<<<launch1(N_SURF,128),128,0,stream>>>(hbuf, 128, sfeat, 16, fA, N_SURF, 160);
  gemm(fA, 160, pp(mlp_w1), pp(mlp_b1), h1, nullptr, 256, N_SURF, 256, 144, 0);
  k_add_ln<<<launch1(N_SURF,128),128,0,stream>>>(h1, nullptr, nullptr, nullptr, nullptr,
                                                 pp(mlp_ln.g), pp(mlp_ln.b), N_SURF, 256, 0, 1);
  k_f32_to_f16<<<launch1((long)N_SURF*256,256),256,0,stream>>>(h1, fA, (long)N_SURF*256);
  gemm(fA, 256, pp(mlp_w2), pp(mlp_b2), hbuf, nullptr, 128, N_SURF, 128, 256, 0);
  k_add_ln<<<launch1(N_SURF,128),128,0,stream>>>(hbuf, nullptr, nullptr, nullptr, nullptr,
                                                 pp(sWv_ln.g), pp(sWv_ln.b), N_SURF, 128, 0, 0);
  k_f32_to_f16<<<launch1((long)N_SURF*128,256),256,0,stream>>>(hbuf, fA, (long)N_SURF*128);
  gemm(fA, 128, pp(sWv.ws_w), pp(sWv.ws_b), hs_s, nullptr, 128, N_SURF, 128, 128, 0);
  k_zero<<<launch1((long)N_SURF*48,256),256,0,stream>>>(hv_s, (long)N_SURF*48);

  // surface edges: svec = pos[s_src]-pos[s_dst]
  k_knn_surf<<<launch1(N_SURF,128),128,0,stream>>>(spos, N_SURF, ssrc);
  k_iota_div<<<launch1(E_SURF,256),256,0,stream>>>(sdst, E_SURF, 16);
  k_edge_geom<<<launch1(E_SURF,256),256,0,stream>>>(spos, ssrc, sdst, E_SURF, raw16, sev);
  k_gvp_edge_We<<<launch1(E_SURF,128),128,0,stream>>>(raw16, sev,
      pp(sWe_ln.b), pp(sWe_ln.g), pp(sWe.wh), pp(sWe.ws_b), pp(sWe.ws_w),
      pp(sWe.wsv_b), pp(sWe.wsv_w), pp(sWe.wv), E_SURF, ses);
  k_zero<<<launch1(N_SURF,256),256,0,stream>>>(cnt_s, N_SURF);
  k_count<<<launch1(E_SURF,256),256,0,stream>>>(sdst, E_SURF, cnt_s);

  // ================= conv layer (shared routine) =================
  auto conv = [&](const ConvP& cp, float* s, float* v, int Nn,
                  const int* src, const int* dst, int E,
                  const float* es, const float* ev, const float* cnt) {
    // --- msg0 ---
    k_vh_msg0<<<launch1((long)E*33,256),256,0,stream>>>(v, ev, src, dst, pp(cp.msg0.wh), E, vh33);
    k_build_msg0A<<<launch1(E,128),128,0,stream>>>(s, es, src, dst, vh33, E, Acat, 352);
    gemm(Acat, 352, pp(cp.msg0.ws_w), pp(cp.msg0.ws_b), ms, ms16, 128, E, 128, 321, 1);
    gemm(ms16, 128, pp(cp.msg0.wsv_w), pp(cp.msg0.wsv_b), gateE, nullptr, 16, E, 16, 128, 0);
    k_vout<<<launch1((long)E*16,256),256,0,stream>>>(vh33, 33, pp(cp.msg0.wv), 16, gateE, E, mv);
    // --- msg1 ---
    k_vh_dense<<<launch1((long)E*16,256),256,0,stream>>>(mv, E, 16, pp(cp.msg1.wh), 16, vh16);
    k_build_cat_vn<<<launch1(E,128),128,0,stream>>>(ms, 128, vh16, 16, Acat, E, 160);
    gemm(Acat, 160, pp(cp.msg1.ws_w), pp(cp.msg1.ws_b), ms, ms16, 128, E, 128, 144, 1);
    gemm(ms16, 128, pp(cp.msg1.wsv_w), pp(cp.msg1.wsv_b), gateE, nullptr, 16, E, 16, 128, 0);
    k_vout<<<launch1((long)E*16,256),256,0,stream>>>(vh16, 16, pp(cp.msg1.wv), 16, gateE, E, mv);
    // --- msg2 (no act) ---
    k_vh_dense<<<launch1((long)E*16,256),256,0,stream>>>(mv, E, 16, pp(cp.msg2.wh), 16, vh16);
    k_build_cat_vn<<<launch1(E,128),128,0,stream>>>(ms, 128, vh16, 16, Acat, E, 160);
    gemm(Acat, 160, pp(cp.msg2.ws_w), pp(cp.msg2.ws_b), ms, ms16, 128, E, 128, 144, 0);
    gemm(ms16, 128, pp(cp.msg2.wsv_w), pp(cp.msg2.wsv_b), gateE, nullptr, 16, E, 16, 128, 0);
    k_vout<<<launch1((long)E*16,256),256,0,stream>>>(vh16, 16, pp(cp.msg2.wv), 16, gateE, E, mv);
    // --- aggregate + ln0 ---
    k_zero<<<launch1((long)Nn*128,256),256,0,stream>>>(ag_s, (long)Nn*128);
    k_zero<<<launch1((long)Nn*48,256),256,0,stream>>>(ag_v, (long)Nn*48);
    k_seg_add<<<launch1((long)E*128,256),256,0,stream>>>(ms, dst, E, 128, ag_s);
    k_seg_add<<<launch1((long)E*48,256),256,0,stream>>>(mv, dst, E, 48, ag_v);
    k_add_ln<<<launch1(Nn,128),128,0,stream>>>(s, v, ag_s, ag_v, cnt,
                                               pp(cp.ln0.g), pp(cp.ln0.b), Nn, 128, 16, 0);
    // --- ff0 ---
    k_vh_dense<<<launch1((long)Nn*32,256),256,0,stream>>>(v, Nn, 16, pp(cp.ff0.wh), 32, vhf);
    k_build_cat_vn<<<launch1(Nn,128),128,0,stream>>>(s, 128, vhf, 32, fA, Nn, 160);
    gemm(fA, 160, pp(cp.ff0.ws_w), pp(cp.ff0.ws_b), fs, fs16, 512, Nn, 512, 160, 1);
    gemm(fs16, 512, pp(cp.ff0.wsv_w), pp(cp.ff0.wsv_b), gateN, nullptr, 32, Nn, 32, 512, 0);
    k_vout<<<launch1((long)Nn*32,256),256,0,stream>>>(vhf, 32, pp(cp.ff0.wv), 32, gateN, Nn, fv);
    // --- ff1 ---
    k_vh_dense<<<launch1((long)Nn*32,256),256,0,stream>>>(fv, Nn, 32, pp(cp.ff1.wh), 32, vhf);
    k_build_cat_vn<<<launch1(Nn,128),128,0,stream>>>(fs, 512, vhf, 32, fA, Nn, 544);
    gemm(fA, 544, pp(cp.ff1.ws_w), pp(cp.ff1.ws_b), hs2, hs2_16, 128, Nn, 128, 544, 0);
    gemm(hs2_16, 128, pp(cp.ff1.wsv_w), pp(cp.ff1.wsv_b), gateN, nullptr, 16, Nn, 16, 128, 0);
    k_vout<<<launch1((long)Nn*16,256),256,0,stream>>>(vhf, 32, pp(cp.ff1.wv), 16, gateN, Nn, hv2);
    // --- residual + ln1 ---
    k_add_ln<<<launch1(Nn,128),128,0,stream>>>(s, v, hs2, hv2, nullptr,
                                               pp(cp.ln1.g), pp(cp.ln1.b), Nn, 128, 16, 0);
  };

  for (int l = 0; l < 3; ++l) {
    conv(layers[l],  s_res, v_res, N_RES,  e_src, e_dst, E_RES,  es_r, ev_r, cnt_r);
    conv(slayers[l], hs_s,  hv_s,  N_SURF, ssrc,  sdst,  E_SURF, ses,  sev,  cnt_s);
  }

  // ================= output heads =================
  k_add_ln<<<launch1(N_RES,128),128,0,stream>>>(s_res, v_res, nullptr, nullptr, nullptr,
                                                pp(Wout_ln.g), pp(Wout_ln.b), N_RES, 128, 16, 0);
  k_vh_dense<<<launch1((long)N_RES*16,256),256,0,stream>>>(v_res, N_RES, 16, pp(Wout.wh), 16, vhf);
  k_build_cat_vn<<<launch1(N_RES,128),128,0,stream>>>(s_res, 128, vhf, 16, fA, N_RES, 160);
  gemm(fA, 160, pp(Wout.ws_w), pp(Wout.ws_b), bbuf, nullptr, 128, N_RES, 128, 144, 1);

  k_add_ln<<<launch1(N_SURF,128),128,0,stream>>>(hs_s, hv_s, nullptr, nullptr, nullptr,
                                                 pp(sWout_ln.g), pp(sWout_ln.b), N_SURF, 128, 16, 0);
  k_vh_dense<<<launch1((long)N_SURF*16,256),256,0,stream>>>(hv_s, N_SURF, 16, pp(sWout.wh), 16, vhf);
  k_build_cat_vn<<<launch1(N_SURF,128),128,0,stream>>>(hs_s, 128, vhf, 16, fA, N_SURF, 160);
  gemm(fA, 160, pp(sWout.ws_w), pp(sWout.ws_b), soutb, nullptr, 128, N_SURF, 128, 144, 1);

  k_final_nf<<<launch1((long)N_RES*128,256),256,0,stream>>>(bbuf, soutb, r2s, out);
  k_graph_sum<<<1,128,0,stream>>>(out);
}
